// Llama3MHAttention_85066122264685
// MI455X (gfx1250) — compile-verified, matrix operand order checked by reading, not run on hardware
//
#include <hip/hip_runtime.h>

#define B_   2
#define T_   2048
#define E_   4096
#define H_   32
#define HKV_ 8
#define DH_  128

typedef __attribute__((ext_vector_type(8)))  unsigned short u16x8;
typedef __attribute__((ext_vector_type(16))) __bf16         bf16x16;
typedef __attribute__((ext_vector_type(8)))  float          f32x8;
typedef __attribute__((ext_vector_type(4)))  unsigned int   u32x4;
typedef __attribute__((ext_vector_type(8)))  int            i32x8;
typedef __attribute__((ext_vector_type(4)))  int            i32x4;

#if defined(__has_builtin)
#  if __has_builtin(__builtin_amdgcn_tensor_load_to_lds)
#    define HAVE_TDM 1
#  endif
#endif
#ifndef HAVE_TDM
#  define HAVE_TDM 0
#endif

// ---- helpers -------------------------------------------------------------

static __device__ __forceinline__ unsigned short f2bf(float f) {
  unsigned int u = __float_as_uint(f);
  u += 0x7FFFu + ((u >> 16) & 1u);          // round-to-nearest-even
  return (unsigned short)(u >> 16);
}

// Build a 16x(bf16) WMMA fragment from two contiguous 8-half (16B) segments.
// CDNA5 16-bit A/B layout: lane L -> row = L%16, half = L/16;
// elements 0..7 = K(half*8 + e), elements 8..15 = K(16 + half*8 + e).
static __device__ __forceinline__ bf16x16 frag16(const unsigned short* p0,
                                                 const unsigned short* p1) {
  union { u16x8 h[2]; bf16x16 v; } u;
  u.h[0] = *(const u16x8*)p0;
  u.h[1] = *(const u16x8*)p1;
  return u.v;
}

#if HAVE_TDM
// Issue one TDM 2D tile load (bf16 elements) global -> LDS.
// Tile: tile_k x tile_rows out of a tensor tensor_k x tensor_rows (row stride
// = tensor_k elements). LDS destination gets 4 DWORDs of padding every
// 16 DWORDs (pad_interval=3, pad_amount=3) -> padded row stride of 40 halves,
// matching the WMMA fragment loader below.
static __device__ __forceinline__ void tdm_load_tile(
    unsigned lds_off, const unsigned short* gptr,
    unsigned tile_k, unsigned tile_rows,
    unsigned tensor_k, unsigned tensor_rows) {
  unsigned long long ga = (unsigned long long)gptr;
  u32x4 g0;
  g0[0] = 1u;                                         // count=1 (valid D#)
  g0[1] = lds_off;                                    // lds_addr (bytes)
  g0[2] = (unsigned)(ga & 0xffffffffu);               // global_addr[31:0]
  g0[3] = (unsigned)((ga >> 32) & 0x1ffffffu)         // global_addr[56:32]
        | (2u << 30);                                 // type = 2 ("image")
  i32x8 g1;
  g1[0] = (1 << 16)                                   // data_size = 2 bytes
        | (1 << 20)                                   // pad_enable
        | (3 << 22)                                   // pad_interval: 16 DW
        | (3 << 25);                                  // pad_amount: 4 DW
  g1[1] = (int)((tensor_k & 0xffffu) << 16);          // tensor_dim0[15:0]
  g1[2] = (int)((tensor_k >> 16) & 0xffffu)           // tensor_dim0[31:16]
        | (int)((tensor_rows & 0xffffu) << 16);       // tensor_dim1[15:0]
  g1[3] = (int)((tensor_rows >> 16) & 0xffffu)        // tensor_dim1[31:16]
        | (int)(tile_k << 16);                        // tile_dim0
  g1[4] = (int)(tile_rows & 0xffffu);                 // tile_dim1 (dim2 = 0)
  g1[5] = (int)tensor_k;                              // tensor_dim0_stride
  g1[6] = 0;
  g1[7] = 0;
  i32x4 z4 = {0, 0, 0, 0};
  i32x8 z8 = {0, 0, 0, 0, 0, 0, 0, 0};
  __builtin_amdgcn_tensor_load_to_lds(g0, g1, z4, z4, z8, 0);
}
#endif

// ---- fp32 -> bf16 cast ---------------------------------------------------

__global__ void cvt_f32_bf16(const float* __restrict__ in,
                             unsigned short* __restrict__ out, int n) {
  int i = blockIdx.x * blockDim.x + threadIdx.x;
  if (i < n) out[i] = f2bf(in[i]);
}

// ---- WMMA GEMM: C[M,N] (f32) = A[M,K](bf16) * B[N,K](bf16)^T -------------
// Block: 128 threads (4 waves). Tile: 128(M) x 64(N), BK=32.
// Wave w owns rows [w*32, w*32+32): 2 A-frags x 4 shared B-frags = 8 WMMAs
// per K-step. Tiles staged to LDS by the Tensor Data Mover (double-buffered,
// TENSORcnt-synchronized); padded LDS stride 40 halves avoids bank conflicts.

__global__ __launch_bounds__(128)
void gemm_bf16_nt(const unsigned short* __restrict__ A,
                  const unsigned short* __restrict__ Bw,
                  float* __restrict__ C, int M, int N, int K) {
  __shared__ unsigned short As[2][128 * 40];
  __shared__ unsigned short Bs[2][64 * 40];
  const int tid  = threadIdx.x;
  const int wv   = tid >> 5;
  const int lane = tid & 31;
  const int lm   = lane & 15;
  const int lh   = (lane >> 4) * 8;        // half-offset in halves (0 or 8)
  const int m0 = blockIdx.y * 128;
  const int n0 = blockIdx.x * 64;

  f32x8 acc[2][4] = {};

#if HAVE_TDM
  const unsigned lds_a0 = (unsigned)(unsigned long long)&As[0][0];
  const unsigned lds_a1 = (unsigned)(unsigned long long)&As[1][0];
  const unsigned lds_b0 = (unsigned)(unsigned long long)&Bs[0][0];
  const unsigned lds_b1 = (unsigned)(unsigned long long)&Bs[1][0];
  if (wv == 0) {   // prologue: tile 0 -> buffer 0
    tdm_load_tile(lds_a0, A + (size_t)m0 * K, 32u, 128u, (unsigned)K, (unsigned)M);
    tdm_load_tile(lds_b0, Bw + (size_t)n0 * K, 32u, 64u, (unsigned)K, (unsigned)N);
  }
#else
  const int lrow = tid >> 2;               // 0..31
  const int lseg = (tid & 3) * 8;          // 0,8,16,24 halves
#endif

  int buf = 0;
  for (int k0 = 0; k0 < K; k0 += 32) {
#if HAVE_TDM
    if (wv == 0) {
      if (k0 + 32 < K) {   // issue next tile into the other buffer
        tdm_load_tile(buf ? lds_a0 : lds_a1,
                      A + (size_t)m0 * K + (k0 + 32), 32u, 128u,
                      (unsigned)K, (unsigned)M);
        tdm_load_tile(buf ? lds_b0 : lds_b1,
                      Bw + (size_t)n0 * K + (k0 + 32), 32u, 64u,
                      (unsigned)K, (unsigned)N);
        __builtin_amdgcn_s_wait_tensorcnt(2);  // current tile (2 older ops) done
      } else {
        __builtin_amdgcn_s_wait_tensorcnt(0);
      }
    }
#else
#pragma unroll
    for (int rr = 0; rr < 128; rr += 32)
      *(u16x8*)&As[buf][(rr + lrow) * 40 + lseg] =
          *(const u16x8*)(A + (size_t)(m0 + rr + lrow) * K + k0 + lseg);
#pragma unroll
    for (int rr = 0; rr < 64; rr += 32)
      *(u16x8*)&Bs[buf][(rr + lrow) * 40 + lseg] =
          *(const u16x8*)(Bw + (size_t)(n0 + rr + lrow) * K + k0 + lseg);
#endif
    __syncthreads();
    const unsigned short* Ab = &As[buf][0];
    const unsigned short* Bb = &Bs[buf][0];
    bf16x16 bfr[4];
#pragma unroll
    for (int nt = 0; nt < 4; ++nt) {
      const int bn = nt * 16 + lm;
      bfr[nt] = frag16(&Bb[bn * 40 + lh], &Bb[bn * 40 + 16 + lh]);
    }
#pragma unroll
    for (int sub = 0; sub < 2; ++sub) {
      const int am = wv * 32 + sub * 16 + lm;
      bf16x16 af = frag16(&Ab[am * 40 + lh], &Ab[am * 40 + 16 + lh]);
#pragma unroll
      for (int nt = 0; nt < 4; ++nt)
        acc[sub][nt] = __builtin_amdgcn_wmma_f32_16x16x32_bf16(
            false, af, false, bfr[nt], (short)0, acc[sub][nt], false, false);
    }
    __syncthreads();
    buf ^= 1;
  }
  // D layout: lane -> n = lm, rows m = (lane/16)*8 + r
#pragma unroll
  for (int sub = 0; sub < 2; ++sub) {
    const int mg = m0 + wv * 32 + sub * 16 + (lane >> 4) * 8;
#pragma unroll
    for (int nt = 0; nt < 4; ++nt) {
      const int ng = n0 + nt * 16 + lm;
#pragma unroll
      for (int r = 0; r < 8; ++r)
        C[(size_t)(mg + r) * N + ng] = acc[sub][nt][r];
    }
  }
}

// ---- RoPE (faithful llama3 scaling) + repack to [B, nh, T, DH] bf16 ------

static __device__ __forceinline__ float rope_inv_freq(int i) {
  float ex  = (2.0f * (float)i) / 128.0f;
  float inv = __expf(-ex * __logf(500000.0f));        // theta^(-2i/DH)
  float wav = 6.28318530717958647f / inv;
  if (wav > 8192.0f) inv *= (1.0f / 32.0f);           // wavelen > cut_hi
  float smooth   = (8192.0f / wav - 1.0f) * (1.0f / 3.0f);
  float smoothed = (1.0f - smooth) * inv * (1.0f / 32.0f) + smooth * inv;
  bool med = (!(wav < 8192.0f)) && (!(wav > 2048.0f)); // exactly as source
  return med ? smoothed : inv;
}

__global__ void rope_pack(const float* __restrict__ Xf,
                          unsigned short* __restrict__ Xb, int nh) {
  int idx = blockIdx.x * blockDim.x + threadIdx.x;  // (b, h, t, i) i fastest
  if (idx >= B_ * nh * T_ * (DH_ / 2)) return;
  int i = idx & 63;
  int r = idx >> 6;
  int t = r & (T_ - 1); r >>= 11;
  int h = r % nh;
  int b = r / nh;
  float inv = rope_inv_freq(i);
  float ang = (float)t * inv;
  float s, c;
  __sincosf(ang, &s, &c);
  const float* row = Xf + (size_t)(b * T_ + t) * (nh * DH_) + h * DH_;
  float x1 = row[i], x2 = row[i + 64];
  unsigned short* orow = Xb + ((size_t)(b * nh + h) * T_ + t) * DH_;
  orow[i]      = f2bf(x1 * c - x2 * s);
  orow[i + 64] = f2bf(x2 * c + x1 * s);
}

// ---- V: [B,T,HKV,DH] f32 -> V^T [B,HKV,DH,T] bf16 ------------------------

__global__ void v_pack_t(const float* __restrict__ Vf,
                         unsigned short* __restrict__ Vt) {
  int idx = blockIdx.x * blockDim.x + threadIdx.x;   // (b,t,h,dh), dh fastest
  if (idx >= B_ * T_ * HKV_ * DH_) return;
  int dh = idx & 127;
  int h  = (idx >> 7) & 7;
  int t  = (idx >> 10) & (T_ - 1);
  int b  = idx >> 21;
  Vt[((size_t)(b * HKV_ + h) * DH_ + dh) * T_ + t] = f2bf(Vf[idx]);
}

// ---- Flash attention: one wave per 16-query tile per head ----------------
// Q [B,H,T,DH] bf16, K [B,HKV,T,DH] bf16, V^T [B,HKV,DH,T] bf16.
// Ctx written bf16 as [B, T, H*DH] (ready to be X of the Wo GEMM).

__global__ __launch_bounds__(32)
void attn_fwd(const unsigned short* __restrict__ Qb,
              const unsigned short* __restrict__ Kb,
              const unsigned short* __restrict__ Vt,
              unsigned short* __restrict__ Ctx) {
  __shared__ unsigned short Pl[16 * 32];   // P tile staging (D -> A layout)
  const int lane = threadIdx.x;
  const int lm = lane & 15;
  const int lh = (lane >> 4) * 8;
  const int q0 = blockIdx.x * 16;
  const int h  = blockIdx.y;
  const int b  = blockIdx.z;
  const int hk = h >> 2;                   // GQA group of 4

  const unsigned short* qrow = Qb + ((size_t)(b * H_ + h) * T_ + q0 + lm) * DH_;
  bf16x16 aq[4];
#pragma unroll
  for (int c = 0; c < 4; ++c)
    aq[c] = frag16(qrow + c * 32 + lh, qrow + c * 32 + 16 + lh);

  f32x8 o[8] = {};
  float mrun[8], lrun[8];
#pragma unroll
  for (int r = 0; r < 8; ++r) { mrun[r] = -__builtin_inff(); lrun[r] = 0.0f; }

  const unsigned short* kbase = Kb + (size_t)(b * HKV_ + hk) * T_ * DH_;
  const unsigned short* vbase = Vt + (size_t)(b * HKV_ + hk) * DH_ * T_;
  const int qr0 = q0 + (lane >> 4) * 8;
  const float sc = 0.08838834764831845f;   // 1/sqrt(128)

  for (int kb = 0; kb < q0 + 16; kb += 32) {
    f32x8 s0 = {}, s1 = {};
#pragma unroll
    for (int c = 0; c < 4; ++c) {          // S0 = Q . K[kb..kb+16)^T
      const unsigned short* kr = kbase + (size_t)(kb + lm) * DH_ + c * 32;
      bf16x16 bk = frag16(kr + lh, kr + 16 + lh);
      s0 = __builtin_amdgcn_wmma_f32_16x16x32_bf16(false, aq[c], false, bk,
                                                   (short)0, s0, false, false);
    }
#pragma unroll
    for (int c = 0; c < 4; ++c) {          // S1 = Q . K[kb+16..kb+32)^T
      const unsigned short* kr = kbase + (size_t)(kb + 16 + lm) * DH_ + c * 32;
      bf16x16 bk = frag16(kr + lh, kr + 16 + lh);
      s1 = __builtin_amdgcn_wmma_f32_16x16x32_bf16(false, aq[c], false, bk,
                                                   (short)0, s1, false, false);
    }
    // scale + causal mask (lane = key column, vgpr r = query row)
    float rmax[8];
#pragma unroll
    for (int r = 0; r < 8; ++r) {
      float a0 = s0[r] * sc;
      float a1 = s1[r] * sc;
      if (kb + lm      > qr0 + r) a0 = -__builtin_inff();
      if (kb + 16 + lm > qr0 + r) a1 = -__builtin_inff();
      s0[r] = a0; s1[r] = a1;
      rmax[r] = fmaxf(a0, a1);
    }
#pragma unroll
    for (int msk = 8; msk >= 1; msk >>= 1)
#pragma unroll
      for (int r = 0; r < 8; ++r)
        rmax[r] = fmaxf(rmax[r], __shfl_xor(rmax[r], msk, 32));
    float alpha[8], rsum[8];
#pragma unroll
    for (int r = 0; r < 8; ++r) {
      float mnew = fmaxf(mrun[r], rmax[r]);
      alpha[r] = __expf(mrun[r] - mnew);
      mrun[r] = mnew;
      float p0 = __expf(s0[r] - mnew);
      float p1 = __expf(s1[r] - mnew);
      s0[r] = p0; s1[r] = p1;
      rsum[r] = p0 + p1;
    }
#pragma unroll
    for (int msk = 8; msk >= 1; msk >>= 1)
#pragma unroll
      for (int r = 0; r < 8; ++r)
        rsum[r] += __shfl_xor(rsum[r], msk, 32);
#pragma unroll
    for (int r = 0; r < 8; ++r)
      lrun[r] = lrun[r] * alpha[r] + rsum[r];
#pragma unroll
    for (int t8 = 0; t8 < 8; ++t8)
#pragma unroll
      for (int r = 0; r < 8; ++r)
        o[t8][r] *= alpha[r];
    // Re-distribute P (D layout) into A layout through LDS
#pragma unroll
    for (int r = 0; r < 8; ++r) {
      int m = (lane >> 4) * 8 + r;
      Pl[m * 32 + lm]      = f2bf(s0[r]);
      Pl[m * 32 + 16 + lm] = f2bf(s1[r]);
    }
    __syncthreads();
    bf16x16 ap = frag16(&Pl[lm * 32 + lh], &Pl[lm * 32 + 16 + lh]);
    __syncthreads();
#pragma unroll
    for (int nt = 0; nt < 8; ++nt) {       // O += P . V  (V^T rows contiguous)
      const unsigned short* vr = vbase + (size_t)(nt * 16 + lm) * T_ + kb;
      bf16x16 bv = frag16(vr + lh, vr + 16 + lh);
      o[nt] = __builtin_amdgcn_wmma_f32_16x16x32_bf16(false, ap, false, bv,
                                                      (short)0, o[nt], false, false);
    }
  }
  float invl[8];
#pragma unroll
  for (int r = 0; r < 8; ++r) invl[r] = 1.0f / lrun[r];
#pragma unroll
  for (int nt = 0; nt < 8; ++nt)
#pragma unroll
    for (int r = 0; r < 8; ++r)
      Ctx[(size_t)(b * T_ + q0 + (lane >> 4) * 8 + r) * E_ + h * DH_ +
          nt * 16 + lm] = f2bf(o[nt][r] * invl[r]);
}

// ---- host launcher -------------------------------------------------------

extern "C" void kernel_launch(void* const* d_in, const int* in_sizes, int n_in,
                              void* d_out, int out_size, void* d_ws, size_t ws_size,
                              hipStream_t stream) {
  (void)in_sizes; (void)n_in; (void)out_size; (void)ws_size;
  const float* x  = (const float*)d_in[0];
  const float* Wq = (const float*)d_in[1];
  const float* Wk = (const float*)d_in[2];
  const float* Wv = (const float*)d_in[3];
  const float* Wo = (const float*)d_in[4];
  float* out = (float*)d_out;
  char* ws = (char*)d_ws;

  const size_t MT = (size_t)B_ * T_;        // 4096 rows
  const size_t KV = (size_t)HKV_ * DH_;     // 1024

  size_t off = 0;
  auto take = [&](size_t bytes) { size_t o = off; off += (bytes + 255) & ~(size_t)255; return o; };
  size_t o_xb  = take(MT * E_ * 2);
  size_t o_wqb = take((size_t)E_ * E_ * 2);
  size_t o_wkb = take(KV * E_ * 2);
  size_t o_wvb = take(KV * E_ * 2);
  size_t o_wob = take((size_t)E_ * E_ * 2);
  size_t o_qf  = take(MT * E_ * 4);
  size_t o_kf  = take(MT * KV * 4);
  size_t o_vf  = take(MT * KV * 4);
  size_t o_qb  = take(MT * E_ * 2);
  size_t o_kb  = take(MT * KV * 2);
  size_t o_vt  = take(MT * KV * 2);
  size_t o_ctx = o_qf;                      // reuse Qf region (Qf dead by then)

  unsigned short* xb   = (unsigned short*)(ws + o_xb);
  unsigned short* wqb  = (unsigned short*)(ws + o_wqb);
  unsigned short* wkb  = (unsigned short*)(ws + o_wkb);
  unsigned short* wvb  = (unsigned short*)(ws + o_wvb);
  unsigned short* wob  = (unsigned short*)(ws + o_wob);
  float*          qf   = (float*)(ws + o_qf);
  float*          kf   = (float*)(ws + o_kf);
  float*          vf   = (float*)(ws + o_vf);
  unsigned short* qb   = (unsigned short*)(ws + o_qb);
  unsigned short* kbuf = (unsigned short*)(ws + o_kb);
  unsigned short* vt   = (unsigned short*)(ws + o_vt);
  unsigned short* ctxb = (unsigned short*)(ws + o_ctx);

  // 1) casts to bf16
  cvt_f32_bf16<<<(int)((MT * E_ + 255) / 256), 256, 0, stream>>>(x, xb, (int)(MT * E_));
  cvt_f32_bf16<<<(int)(((size_t)E_ * E_ + 255) / 256), 256, 0, stream>>>(Wq, wqb, E_ * E_);
  cvt_f32_bf16<<<(int)((KV * E_ + 255) / 256), 256, 0, stream>>>(Wk, wkb, (int)(KV * E_));
  cvt_f32_bf16<<<(int)((KV * E_ + 255) / 256), 256, 0, stream>>>(Wv, wvb, (int)(KV * E_));
  cvt_f32_bf16<<<(int)(((size_t)E_ * E_ + 255) / 256), 256, 0, stream>>>(Wo, wob, E_ * E_);

  // 2) QKV projections (WMMA GEMMs, TDM-fed, 128x64 block tiles)
  gemm_bf16_nt<<<dim3(E_ / 64, (int)(MT / 128)), 128, 0, stream>>>(xb, wqb, qf, (int)MT, E_, E_);
  gemm_bf16_nt<<<dim3((int)(KV / 64), (int)(MT / 128)), 128, 0, stream>>>(xb, wkb, kf, (int)MT, (int)KV, E_);
  gemm_bf16_nt<<<dim3((int)(KV / 64), (int)(MT / 128)), 128, 0, stream>>>(xb, wvb, vf, (int)MT, (int)KV, E_);

  // 3) RoPE + layout packs
  rope_pack<<<(B_ * H_ * T_ * (DH_ / 2)) / 256, 256, 0, stream>>>(qf, qb, H_);
  rope_pack<<<(B_ * HKV_ * T_ * (DH_ / 2)) / 256, 256, 0, stream>>>(kf, kbuf, HKV_);
  v_pack_t<<<(B_ * T_ * HKV_ * DH_) / 256, 256, 0, stream>>>(vf, vt);

  // 4) flash attention (WMMA)
  attn_fwd<<<dim3(T_ / 16, H_, B_), 32, 0, stream>>>(qb, kbuf, vt, ctxb);

  // 5) output projection straight into d_out (fp32)
  gemm_bf16_nt<<<dim3(E_ / 64, (int)(MT / 128)), 128, 0, stream>>>(ctxb, wob, out, (int)MT, E_, E_);
}
